// PointNetFeaturePropagation_30391188587133
// MI455X (gfx1250) — matrix operand order repaired and checked by reading, not compiled, not run on hardware
//
#include <hip/hip_runtime.h>
#include <hip/hip_bf16.h>

// ---------- problem constants ----------
constexpr int B_   = 4;
constexpr int N1_  = 16384;
constexpr int N2_  = 4096;
constexpr int C1_  = 128;
constexpr int C2_  = 256;
constexpr int K1_  = 384;            // C1 + C2
constexpr int M1_  = 256;            // layer-1 out channels
constexpr int M2_  = 128;            // layer-2 out channels
constexpr int NCOLS_ = B_ * N1_;     // 65536 GEMM columns

typedef __attribute__((ext_vector_type(16))) _Float16 v16h;
typedef __attribute__((ext_vector_type(8)))  _Float16 v8h;
typedef __attribute__((ext_vector_type(8)))  float    v8f;
typedef __attribute__((ext_vector_type(4)))  float    v4f;

// =====================================================================
// 1) 3-NN search: one thread per query point, xyz2 staged in LDS tiles
// =====================================================================
__global__ __launch_bounds__(256) void knn_kernel(const float* __restrict__ xyz1,
                                                  const float* __restrict__ xyz2,
                                                  int* __restrict__ idx_out,
                                                  float* __restrict__ w_out)
{
    __shared__ float sx[1024], sy[1024], sz[1024];
    const int per_b = N1_ / 256;                 // 64 blocks per batch
    const int b = blockIdx.x / per_b;
    const int q = (blockIdx.x % per_b) * 256 + threadIdx.x;

    const float* x1 = xyz1 + (size_t)b * 3 * N1_;
    const float qx = x1[q];
    const float qy = x1[N1_ + q];
    const float qz = x1[2 * N1_ + q];

    float d0 = 3e38f, d1 = 3e38f, d2 = 3e38f;
    int   i0 = 0, i1 = 0, i2 = 0;

    const float* x2 = xyz2 + (size_t)b * 3 * N2_;
    for (int t0 = 0; t0 < N2_; t0 += 1024) {
        __syncthreads();
        for (int i = threadIdx.x; i < 1024; i += 256) {
            sx[i] = x2[t0 + i];
            sy[i] = x2[N2_ + t0 + i];
            sz[i] = x2[2 * N2_ + t0 + i];
        }
        __syncthreads();
        for (int i = 0; i < 1024; ++i) {
            float dx = qx - sx[i], dy = qy - sy[i], dz = qz - sz[i];
            float d = dx * dx + dy * dy + dz * dz;
            int gi = t0 + i;
            if (d < d2) {
                if (d < d1) {
                    if (d < d0) { d2 = d1; i2 = i1; d1 = d0; i1 = i0; d0 = d; i0 = gi; }
                    else        { d2 = d1; i2 = i1; d1 = d;  i1 = gi; }
                } else          { d2 = d;  i2 = gi; }
            }
        }
    }
    d0 = fmaxf(d0, 1e-10f); d1 = fmaxf(d1, 1e-10f); d2 = fmaxf(d2, 1e-10f);
    float w0 = 1.f / d0, w1 = 1.f / d1, w2 = 1.f / d2;
    float inv = 1.f / (w0 + w1 + w2);
    size_t base = ((size_t)b * N1_ + q) * 3;
    idx_out[base] = i0; idx_out[base + 1] = i1; idx_out[base + 2] = i2;
    w_out[base] = w0 * inv; w_out[base + 1] = w1 * inv; w_out[base + 2] = w2 * inv;
}

// =====================================================================
// 2) pack weights f32 -> f16 (row-major, Cin contiguous)
// =====================================================================
__global__ __launch_bounds__(256) void pack_weights(const float* __restrict__ W1,
                                                    const float* __restrict__ W2,
                                                    _Float16* __restrict__ W1p,
                                                    _Float16* __restrict__ W2p)
{
    int t = blockIdx.x * 256 + threadIdx.x;
    const int n1 = M1_ * K1_;       // 98304
    const int n2 = M2_ * M1_;       // 32768
    if (t < n1)              W1p[t] = (_Float16)W1[t];
    else if (t < n1 + n2)    W2p[t - n1] = (_Float16)W2[t - n1];
}

// =====================================================================
// 3) build X (K1 x NCOLS, f16, column-major: K contiguous per column)
//    rows [0,128)   = points1
//    rows [128,384) = 3-NN interpolated points2
// =====================================================================
__global__ __launch_bounds__(256) void build_x_p1(const float* __restrict__ p1,
                                                  _Float16* __restrict__ X)
{
    size_t t = (size_t)blockIdx.x * 256 + threadIdx.x;   // < NCOLS*C1
    int    c = (int)(t & (C1_ - 1));
    size_t j = t >> 7;                                   // column index
    int    b = (int)(j >> 14);
    int    n = (int)(j & (N1_ - 1));
    X[j * K1_ + c] = (_Float16)p1[((size_t)b * C1_ + c) * N1_ + n];
}

__global__ __launch_bounds__(256) void build_x_p2(const float* __restrict__ p2,
                                                  const int* __restrict__ nidx,
                                                  const float* __restrict__ nw,
                                                  _Float16* __restrict__ X)
{
    size_t t = (size_t)blockIdx.x * 256 + threadIdx.x;   // < NCOLS*C2
    size_t j = t & (size_t)(NCOLS_ - 1);
    int    c = (int)(t >> 16);
    int    b = (int)(j >> 14);
    const float* row = p2 + ((size_t)b * C2_ + c) * N2_;
    size_t base = j * 3;
    int   i0 = nidx[base], i1 = nidx[base + 1], i2 = nidx[base + 2];
    float w0 = nw[base],   w1 = nw[base + 1],   w2 = nw[base + 2];
    X[j * K1_ + C1_ + c] = (_Float16)(w0 * row[i0] + w1 * row[i1] + w2 * row[i2]);
}

// =====================================================================
// 4) WMMA GEMM: Y(M x NCOLS, f32, col-major) = Wp(MxK) * X(KxNCOLS)
//    + fused per-channel sum / sum-of-squares for BatchNorm stats.
//    Block: 8 waves (2 M x 4 N), wave tile 32x32 (2x2 fragments),
//    k-step 32 via v_wmma_f32_16x16x32_f16.
// =====================================================================
template <int M, int K>
__global__ __launch_bounds__(256) void wmma_gemm(const _Float16* __restrict__ Wp,
                                                 const _Float16* __restrict__ X,
                                                 float* __restrict__ Y,
                                                 float* __restrict__ ssum,
                                                 float* __restrict__ ssq)
{
    constexpr int MB = 64, NB = 128;
    const int wave = threadIdx.x >> 5;
    const int lane = threadIdx.x & 31;
    const int wm = wave >> 2;            // 0..1
    const int wn = wave & 3;             // 0..3
    const int mBlocks = M / MB;
    const int bm = blockIdx.x % mBlocks;
    const int bn = blockIdx.x / mBlocks;
    const int m0 = bm * MB + wm * 32;
    const int n0 = bn * NB + wn * 32;
    const int half = lane >> 4;          // ISA: lanes 16-31 carry the other K half
    const int l16  = lane & 15;

    v8f acc00 = {}, acc01 = {}, acc10 = {}, acc11 = {};

    // A fragment: row = m0 + l16 ; per-lane K chunks {half*8 .. +7} and {+16 .. +23}
    const _Float16* a0p = Wp + (size_t)(m0 + l16) * K + half * 8;
    const _Float16* a1p = a0p + (size_t)16 * K;
    // B fragment: col = n0 + l16 ; per-lane K chunks {half*8 .. +7} and {+16 .. +23}
    const _Float16* b0p = X + (size_t)(n0 + l16) * K + half * 8;
    const _Float16* b1p = b0p + (size_t)16 * K;

#pragma unroll 4
    for (int k = 0; k < K; k += 32) {
        v16h a0, a1, b0, b1;
        ((v8h*)&a0)[0] = *(const v8h*)(a0p + k);
        ((v8h*)&a0)[1] = *(const v8h*)(a0p + k + 16);
        ((v8h*)&a1)[0] = *(const v8h*)(a1p + k);
        ((v8h*)&a1)[1] = *(const v8h*)(a1p + k + 16);
        ((v8h*)&b0)[0] = *(const v8h*)(b0p + k);
        ((v8h*)&b0)[1] = *(const v8h*)(b0p + k + 16);
        ((v8h*)&b1)[0] = *(const v8h*)(b1p + k);
        ((v8h*)&b1)[1] = *(const v8h*)(b1p + k + 16);
        acc00 = __builtin_amdgcn_wmma_f32_16x16x32_f16(false, a0, false, b0, (short)0, acc00, false, false);
        acc01 = __builtin_amdgcn_wmma_f32_16x16x32_f16(false, a0, false, b1, (short)0, acc01, false, false);
        acc10 = __builtin_amdgcn_wmma_f32_16x16x32_f16(false, a1, false, b0, (short)0, acc10, false, false);
        acc11 = __builtin_amdgcn_wmma_f32_16x16x32_f16(false, a1, false, b1, (short)0, acc11, false, false);
    }

    // ---- store D fragments (lane holds 8 consecutive M rows of one column)
    const int col0 = n0 + l16;
    const int col1 = n0 + 16 + l16;
    const int r0   = m0 + half * 8;
    const int r1   = m0 + 16 + half * 8;
    float* y;
    y = Y + (size_t)col0 * M + r0; ((v4f*)y)[0] = ((v4f*)&acc00)[0]; ((v4f*)y)[1] = ((v4f*)&acc00)[1];
    y = Y + (size_t)col1 * M + r0; ((v4f*)y)[0] = ((v4f*)&acc01)[0]; ((v4f*)y)[1] = ((v4f*)&acc01)[1];
    y = Y + (size_t)col0 * M + r1; ((v4f*)y)[0] = ((v4f*)&acc10)[0]; ((v4f*)y)[1] = ((v4f*)&acc10)[1];
    y = Y + (size_t)col1 * M + r1; ((v4f*)y)[0] = ((v4f*)&acc11)[0]; ((v4f*)y)[1] = ((v4f*)&acc11)[1];

    // ---- fused BN statistics: reduce over this wave's 32 columns, atomic per channel
    float s[8], sq[8];
#pragma unroll
    for (int r = 0; r < 8; ++r) {
        float x0 = acc00[r], x1 = acc01[r];
        s[r] = x0 + x1; sq[r] = x0 * x0 + x1 * x1;
    }
    for (int mask = 1; mask < 16; mask <<= 1) {
#pragma unroll
        for (int r = 0; r < 8; ++r) { s[r] += __shfl_xor(s[r], mask); sq[r] += __shfl_xor(sq[r], mask); }
    }
    if (l16 == 0) {
#pragma unroll
        for (int r = 0; r < 8; ++r) {
            atomicAdd(ssum + m0 + half * 8 + r, s[r]);
            atomicAdd(ssq  + m0 + half * 8 + r, sq[r]);
        }
    }
#pragma unroll
    for (int r = 0; r < 8; ++r) {
        float x0 = acc10[r], x1 = acc11[r];
        s[r] = x0 + x1; sq[r] = x0 * x0 + x1 * x1;
    }
    for (int mask = 1; mask < 16; mask <<= 1) {
#pragma unroll
        for (int r = 0; r < 8; ++r) { s[r] += __shfl_xor(s[r], mask); sq[r] += __shfl_xor(sq[r], mask); }
    }
    if (l16 == 0) {
#pragma unroll
        for (int r = 0; r < 8; ++r) {
            atomicAdd(ssum + m0 + 16 + half * 8 + r, s[r]);
            atomicAdd(ssq  + m0 + 16 + half * 8 + r, sq[r]);
        }
    }
}

// =====================================================================
// 5) BN finalize: per-channel scale/shift from batch stats (bias cancels)
// =====================================================================
__global__ void bn_finalize(const float* __restrict__ ssum, const float* __restrict__ ssq,
                            const float* __restrict__ g, const float* __restrict__ be,
                            float* __restrict__ scale, float* __restrict__ shift, int M)
{
    int m = blockIdx.x * blockDim.x + threadIdx.x;
    if (m >= M) return;
    const float inv_n = 1.0f / (float)NCOLS_;
    float mean = ssum[m] * inv_n;
    float var  = ssq[m] * inv_n - mean * mean;
    float sc   = g[m] * rsqrtf(var + 1e-5f);
    scale[m] = sc;
    shift[m] = be[m] - mean * sc;
}

// 6) BN + ReLU -> f16 activations for layer 2 (identity col-major layout)
__global__ __launch_bounds__(256) void bn_relu_f16(const float* __restrict__ Y,
                                                   const float* __restrict__ scale,
                                                   const float* __restrict__ shift,
                                                   _Float16* __restrict__ Xo)
{
    size_t t = (size_t)blockIdx.x * 256 + threadIdx.x;   // < M1*NCOLS
    int c = (int)(t & (size_t)(M1_ - 1));
    float v = Y[t] * scale[c] + shift[c];
    Xo[t] = (_Float16)fmaxf(v, 0.f);
}

// 7) BN + ReLU -> f32 output in (B, 128, N1) layout
__global__ __launch_bounds__(256) void bn_relu_out(const float* __restrict__ Y2,
                                                   const float* __restrict__ scale,
                                                   const float* __restrict__ shift,
                                                   float* __restrict__ out)
{
    size_t t = (size_t)blockIdx.x * 256 + threadIdx.x;   // < B*M2*N1
    int n = (int)(t & (size_t)(N1_ - 1));
    int c = (int)((t >> 14) & (size_t)(M2_ - 1));
    int b = (int)(t >> 21);
    size_t j = (size_t)b * N1_ + n;
    float v = Y2[j * M2_ + c] * scale[c] + shift[c];
    out[t] = fmaxf(v, 0.f);
}

// =====================================================================
extern "C" void kernel_launch(void* const* d_in, const int* in_sizes, int n_in,
                              void* d_out, int out_size, void* d_ws, size_t ws_size,
                              hipStream_t stream)
{
    (void)in_sizes; (void)n_in; (void)out_size; (void)ws_size;
    const float* xyz1    = (const float*)d_in[0];
    const float* xyz2    = (const float*)d_in[1];
    const float* points1 = (const float*)d_in[2];
    const float* points2 = (const float*)d_in[3];
    const float* W1      = (const float*)d_in[4];
    // d_in[5] = b1 : cancels exactly in training-mode BN
    const float* g1      = (const float*)d_in[6];
    const float* be1     = (const float*)d_in[7];
    const float* W2      = (const float*)d_in[8];
    // d_in[9] = b2 : cancels
    const float* g2      = (const float*)d_in[10];
    const float* be2     = (const float*)d_in[11];

    char* ws = (char*)d_ws;
    size_t off = 0;
    auto alloc = [&](size_t bytes) {
        void* p = ws + off;
        off = (off + bytes + 255) & ~(size_t)255;
        return p;
    };
    _Float16* X1  = (_Float16*)alloc((size_t)K1_ * NCOLS_ * 2);   // 48 MB
    float*    Y2  = (float*)X1;                                   // alias: X1 dead before GEMM2 writes (32 MB)
    float*    Y1  = (float*)   alloc((size_t)M1_ * NCOLS_ * 4);   // 64 MB
    _Float16* X2  = (_Float16*)alloc((size_t)M1_ * NCOLS_ * 2);   // 32 MB
    _Float16* W1p = (_Float16*)alloc((size_t)M1_ * K1_ * 2);
    _Float16* W2p = (_Float16*)alloc((size_t)M2_ * M1_ * 2);
    int*      nidx = (int*)    alloc((size_t)B_ * N1_ * 3 * 4);
    float*    nw   = (float*)  alloc((size_t)B_ * N1_ * 3 * 4);
    float*    stats = (float*) alloc(8 * 1024);
    float *sum1 = stats,        *sq1 = stats + 256,  *sc1 = stats + 512,  *sh1 = stats + 768;
    float *sum2 = stats + 1024, *sq2 = stats + 1280, *sc2 = stats + 1536, *sh2 = stats + 1792;

    hipMemsetAsync(stats, 0, 8 * 1024, stream);

    pack_weights<<<512, 256, 0, stream>>>(W1, W2, W1p, W2p);
    knn_kernel<<<B_ * N1_ / 256, 256, 0, stream>>>(xyz1, xyz2, nidx, nw);
    build_x_p1<<<(NCOLS_ / 256) * C1_, 256, 0, stream>>>(points1, X1);
    build_x_p2<<<(NCOLS_ / 256) * C2_, 256, 0, stream>>>(points2, nidx, nw, X1);

    wmma_gemm<M1_, K1_><<<(M1_ / 64) * (NCOLS_ / 128), 256, 0, stream>>>(W1p, X1, Y1, sum1, sq1);
    bn_finalize<<<1, 256, 0, stream>>>(sum1, sq1, g1, be1, sc1, sh1, M1_);
    bn_relu_f16<<<(int)((size_t)M1_ * NCOLS_ / 256), 256, 0, stream>>>(Y1, sc1, sh1, X2);

    wmma_gemm<M2_, M1_><<<(M2_ / 64) * (NCOLS_ / 128), 256, 0, stream>>>(W2p, X2, Y2, sum2, sq2);
    bn_finalize<<<1, 256, 0, stream>>>(sum2, sq2, g2, be2, sc2, sh2, M2_);
    bn_relu_out<<<(int)((size_t)M2_ * NCOLS_ / 256), 256, 0, stream>>>(Y2, sc2, sh2, (float*)d_out);
}
